// ConvNeXtV2KalmanNet_63007170233033
// MI455X (gfx1250) — matrix operand order repaired and checked by reading, not compiled, not run on hardware
//
#include <hip/hip_runtime.h>
#include <math.h>

// ---------------------------------------------------------------------------
// ConvNeXtV2 + KalmanNet + GRU rollout for MI455X (gfx1250, wave32, WMMA).
//
// Shapes: B=256, Q=512, D=32, H=192, HID=384, IN_CH=96, KERNEL=9, W_OUT=64.
//
// Heavy GEMMs run on v_wmma_f32_16x16x32_bf16 (f32 accumulate). Activations
// feeding GEMMs are stored "k-panelized":
//     X[((b*(K/32) + k/32) * T + t) * 32 + (k%32)]
// so every WMMA B-fragment is 16 contiguous bf16 (one b128 pair per lane,
// contiguous 1KB per wave). K/T/O and the fused epilogue are template params
// so k-loops fully unroll and all epilogue addresses fold into immediate
// offsets; each wave computes 2 tiles sharing one operand fragment.
// ---------------------------------------------------------------------------

#define B_    256
#define Q_    512
#define D_    32
#define H_    192
#define HID_  384
#define INCH_ 96
#define KER_  9
#define WOUT_ 64
#define PI_F  3.14159265358979f

typedef __bf16 bf16_t;
typedef __attribute__((ext_vector_type(16))) __bf16 v16bf;
typedef __attribute__((ext_vector_type(8)))  float  v8f;

__device__ __forceinline__ bf16_t f2bf(float f) {
  unsigned u; __builtin_memcpy(&u, &f, 4);
  unsigned r = (u + 0x7FFFu + ((u >> 16) & 1u)) >> 16;   // RNE
  unsigned short s = (unsigned short)r;
  bf16_t b; __builtin_memcpy(&b, &s, 2);
  return b;
}
__device__ __forceinline__ float bf2f(bf16_t b) {
  unsigned short s; __builtin_memcpy(&s, &b, 2);
  unsigned u = ((unsigned)s) << 16;
  float f; __builtin_memcpy(&f, &u, 4);
  return f;
}
__device__ __forceinline__ float sigmoidf_(float x) { return 1.f / (1.f + expf(-x)); }
__device__ __forceinline__ float geluf_(float x) { return 0.5f * x * (1.f + erff(x * 0.70710678118f)); }

// ---------------------------------------------------------------------------
// Weight conversion f32 -> bf16
// ---------------------------------------------------------------------------
__global__ void f32_to_bf16_kernel(const float* __restrict__ s, bf16_t* __restrict__ d, int n) {
  int i = blockIdx.x * blockDim.x + threadIdx.x;
  if (i < n) d[i] = f2bf(s[i]);
}

// ---------------------------------------------------------------------------
// Features: feats panels [x | dy | ddy], bf16, panel layout (B, 3, Q, 32).
// Thread map d-fastest: x_in reads and panel writes both coalesce.
// ---------------------------------------------------------------------------
__global__ void features_kernel(const float* __restrict__ x, bf16_t* __restrict__ feats) {
  int tid = blockIdx.x * blockDim.x + threadIdx.x;   // ((b*512 + t)*32 + d)
  int d = tid & 31;
  int t = (tid >> 5) & (Q_ - 1);
  int b = tid >> 14;
  const float* xr = x + ((size_t)b * Q_ + t) * D_ + d;
  float x0  = xr[0];
  float xm1 = (t >= 1) ? xr[-D_] : 0.f;
  float xm2 = (t >= 2) ? xr[-2 * D_] : 0.f;
  float dy   = (t >= 1) ? (x0 - xm1) : 0.f;
  float dym1 = (t >= 2) ? (xm1 - xm2) : 0.f;
  float ddy  = (t >= 1) ? (dy - dym1) : 0.f;
  size_t base = (((size_t)b * 3) * Q_ + t) * 32 + d;
  feats[base]                        = f2bf(x0);
  feats[base + (size_t)Q_ * 32]      = f2bf(dy);
  feats[base + (size_t)2 * Q_ * 32]  = f2bf(ddy);
}

// ---------------------------------------------------------------------------
// WMMA GEMM #1:  out[b, o, t] = act( sum_k W[o,k] * X_panel[b,k,t] + bias[o] )
//   W: (O,K) bf16 row-major.  X: panel layout (B, K/32, T, 32) bf16.
//   WF/RESID use row-major f32 (B,O,T); WB writes panel layout (B,O/32,T,32).
//   ACT: 0=none, 1=exact GELU.  Fully unrolled; 2 t-tiles per wave share the
//   weight A-fragment.  All fragment loads are contiguous b128s; epilogue
//   addresses are base + constant immediate offsets (r*T folds at compile
//   time since T is a template parameter).
// Grid: (T/128, O/16, B), block 128 (4 waves).
// ---------------------------------------------------------------------------
template <int K, int T, int ACT, bool RESID, bool WF, bool WB>
__global__ void gemm_wx_kernel(const bf16_t* __restrict__ W, const bf16_t* __restrict__ X,
                               const float* __restrict__ bias, const float* __restrict__ resid,
                               float* __restrict__ outf, bf16_t* __restrict__ outb,
                               int O) {
  const int wid  = threadIdx.x >> 5;
  const int lane = threadIdx.x & 31;
  const int half = lane >> 4;
  const int mrow = lane & 15;
  const int bb = blockIdx.z;
  const int o0 = blockIdx.y * 16;
  const int t0 = (blockIdx.x * (blockDim.x >> 5) + wid) * 32;   // two 16-wide tiles

  v8f acc0 = {0.f, 0.f, 0.f, 0.f, 0.f, 0.f, 0.f, 0.f};
  v8f acc1 = {0.f, 0.f, 0.f, 0.f, 0.f, 0.f, 0.f, 0.f};
  const bf16_t* wp = W + (size_t)(o0 + mrow) * K;
  // panel base for (bb, kp=0, t0+mrow): fragment = 16 contiguous bf16 at +half*16
  const bf16_t* xp = X + (((size_t)bb * (K / 32)) * T + (t0 + mrow)) * 32 + half * 16;

  #pragma unroll
  for (int k0 = 0; k0 < K; k0 += 32) {
    v16bf a, b0, b1;
    const bf16_t* pa = wp + k0;
    if (k0 + 32 < K) __builtin_prefetch((const void*)(pa + 32), 0, 1);  // global_prefetch_b8
    // A 16x32 bf16: lanes {0-15,16-31} hold M=mrow; K = half*16 + e (contiguous)
    #pragma unroll
    for (int e = 0; e < 8; ++e) {
      a[e]     = pa[half * 16 + e];
      a[8 + e] = pa[half * 16 + 8 + e];
    }
    // B 32x16 bf16: lane holds column N=mrow; K = half*16 + e  (contiguous panel)
    const bf16_t* pb = xp + (size_t)(k0 >> 5) * T * 32;
    #pragma unroll
    for (int e = 0; e < 16; ++e) {
      b0[e] = pb[e];
      b1[e] = pb[16 * 32 + e];   // column t0+16+mrow
    }
    acc0 = __builtin_amdgcn_wmma_f32_16x16x32_bf16(false, a, false, b0, (short)0, acc0,
                                                   false, false);
    acc1 = __builtin_amdgcn_wmma_f32_16x16x32_bf16(false, a, false, b1, (short)0, acc1,
                                                   false, false);
  }
  // C/D: lane (half,mrow) -> (M = half*8 + r, N = mrow)
  const int ob = o0 + half * 8;                    // first output row of this lane
  float r0[8], r1[8];
  #pragma unroll
  for (int r = 0; r < 8; ++r) {
    float bv = bias[ob + r];
    float v0 = acc0[r] + bv;
    float v1 = acc1[r] + bv;
    if constexpr (ACT == 1) { v0 = geluf_(v0); v1 = geluf_(v1); }
    r0[r] = v0; r1[r] = v1;
  }
  if constexpr (RESID) {
    const float* pr = resid + ((size_t)bb * O + ob) * T + (t0 + mrow);
    #pragma unroll
    for (int r = 0; r < 8; ++r) { r0[r] += pr[r * T]; r1[r] += pr[r * T + 16]; }
  }
  if constexpr (WF) {
    float* po = outf + ((size_t)bb * O + ob) * T + (t0 + mrow);
    #pragma unroll
    for (int r = 0; r < 8; ++r) { po[r * T] = r0[r]; po[r * T + 16] = r1[r]; }
  }
  if constexpr (WB) {   // panel layout; 8 consecutive o -> one 16B store per tile
    bf16_t p0[8], p1[8];
    #pragma unroll
    for (int r = 0; r < 8; ++r) { p0[r] = f2bf(r0[r]); p1[r] = f2bf(r1[r]); }
    size_t base = (((size_t)bb * (O >> 5) + (ob >> 5)) * T + (t0 + mrow)) * 32 + (ob & 31);
    __builtin_memcpy(outb + base, p0, 16);
    __builtin_memcpy(outb + base + (size_t)16 * 32, p1, 16);
  }
}

// ---------------------------------------------------------------------------
// WMMA GEMM #2:  out[m, o] = act( sum_k A[m,k] * W[o,k] + bias[o] )
//   A: (M,K) bf16 row stride lda,  W: (O,K) bf16 row-major (both contiguous).
//   ACT: 0=none, 1=tanh.  2 batch tiles per wave share the weight B-fragment.
//   O is a template parameter so epilogue offsets (r*O) are immediates.
// Grid: (O/64, M/32), block 128.
// ---------------------------------------------------------------------------
template <int K, int O, int ACT, bool WF, bool WB>
__global__ void gemm_bt_kernel(const bf16_t* __restrict__ A, int lda,
                               const bf16_t* __restrict__ W,
                               const float* __restrict__ bias,
                               float* __restrict__ outf, bf16_t* __restrict__ outb) {
  const int wid  = threadIdx.x >> 5;
  const int lane = threadIdx.x & 31;
  const int half = lane >> 4;
  const int mrow = lane & 15;
  const int b0 = blockIdx.y * 32;   // two 16-row batch tiles
  const int o0 = (blockIdx.x * (blockDim.x >> 5) + wid) * 16;

  v8f acc0 = {0.f, 0.f, 0.f, 0.f, 0.f, 0.f, 0.f, 0.f};
  v8f acc1 = {0.f, 0.f, 0.f, 0.f, 0.f, 0.f, 0.f, 0.f};
  const bf16_t* ap0 = A + (size_t)(b0 + mrow) * lda;
  const bf16_t* ap1 = A + (size_t)(b0 + 16 + mrow) * lda;
  const bf16_t* wp  = W + (size_t)(o0 + mrow) * K;

  #pragma unroll
  for (int k0 = 0; k0 < K; k0 += 32) {
    v16bf a0, a1, bfr;
    const bf16_t* pa0 = ap0 + k0;
    const bf16_t* pa1 = ap1 + k0;
    #pragma unroll
    for (int e = 0; e < 8; ++e) {
      a0[e]     = pa0[half * 8 + e];
      a0[8 + e] = pa0[16 + half * 8 + e];
      a1[e]     = pa1[half * 8 + e];
      a1[8 + e] = pa1[16 + half * 8 + e];
    }
    const bf16_t* pb = wp + k0 + half * 8;    // K = half*8+e (e<8) / 16+half*8+(e-8)
    #pragma unroll
    for (int e = 0; e < 16; ++e) bfr[e] = pb[e + (e & 8)];
    acc0 = __builtin_amdgcn_wmma_f32_16x16x32_bf16(false, a0, false, bfr, (short)0, acc0,
                                                   false, false);
    acc1 = __builtin_amdgcn_wmma_f32_16x16x32_bf16(false, a1, false, bfr, (short)0, acc1,
                                                   false, false);
  }
  const int o = o0 + mrow;
  const float bv = bias[o];
  float* pf0; float* pf1; bf16_t* pb0; bf16_t* pb1;
  if constexpr (WF) {
    pf0 = outf + (size_t)(b0 + half * 8) * O + o;
    pf1 = outf + (size_t)(b0 + 16 + half * 8) * O + o;
  }
  if constexpr (WB) {
    pb0 = outb + (size_t)(b0 + half * 8) * O + o;
    pb1 = outb + (size_t)(b0 + 16 + half * 8) * O + o;
  }
  #pragma unroll
  for (int r = 0; r < 8; ++r) {
    float v0 = acc0[r] + bv;
    float v1 = acc1[r] + bv;
    if constexpr (ACT == 1) { v0 = tanhf(v0); v1 = tanhf(v1); }
    if constexpr (WF) { pf0[r * O] = v0; pf1[r * O] = v1; }
    if constexpr (WB) { pb0[r * O] = f2bf(v0); pb1[r * O] = f2bf(v1); }
  }
}

// ---------------------------------------------------------------------------
// Depthwise conv K=9, edge padding, + bias.  h,out: (B,192,512) f32
// ---------------------------------------------------------------------------
__global__ void dwconv_kernel(const float* __restrict__ h, const float* __restrict__ w,
                              const float* __restrict__ bias, float* __restrict__ out) {
  size_t tid = (size_t)blockIdx.x * blockDim.x + threadIdx.x;
  int t = (int)(tid & (Q_ - 1));
  size_t bc = tid >> 9;
  int c = (int)(bc % H_);
  const float* base = h + (bc << 9);
  const float* wr = w + c * KER_;
  float acc = bias[c];
  #pragma unroll
  for (int j = 0; j < KER_; ++j) {
    int tt = t + j - 4;
    tt = tt < 0 ? 0 : (tt > Q_ - 1 ? Q_ - 1 : tt);
    acc += base[tt] * wr[j];
  }
  out[tid] = acc;
}

// ---------------------------------------------------------------------------
// LayerNorm over channel dim (192) per (b,t).
//   _bf variant: bf16 panel layout (B, 6, Q, 32), 64B vector stores per panel.
//   _t  variant: f32 (B, Q, 192) transposed.
// ---------------------------------------------------------------------------
__global__ void ln_ch_bf_kernel(const float* __restrict__ in, bf16_t* __restrict__ out,
                                const float* __restrict__ g, const float* __restrict__ be) {
  int tid = blockIdx.x * blockDim.x + threadIdx.x;
  int t = tid & (Q_ - 1);
  int b = tid >> 9;
  const float* base = in + (size_t)b * H_ * Q_ + t;
  float s = 0.f, s2 = 0.f;
  for (int c = 0; c < H_; ++c) { float v = base[(size_t)c * Q_]; s += v; s2 += v * v; }
  float mean = s * (1.f / H_);
  float var  = s2 * (1.f / H_) - mean * mean;
  float inv  = rsqrtf(var + 1e-5f);
  #pragma unroll
  for (int p = 0; p < H_ / 32; ++p) {
    bf16_t tmp[32];
    #pragma unroll
    for (int ci = 0; ci < 32; ++ci) {
      int c = p * 32 + ci;
      tmp[ci] = f2bf((base[(size_t)c * Q_] - mean) * inv * g[c] + be[c]);
    }
    __builtin_memcpy(out + (((size_t)b * (H_ / 32) + p) * Q_ + t) * 32, tmp, 64);
  }
}

__global__ void ln_ch_t_kernel(const float* __restrict__ in, float* __restrict__ out,
                               const float* __restrict__ g, const float* __restrict__ be) {
  int tid = blockIdx.x * blockDim.x + threadIdx.x;
  int t = tid & (Q_ - 1);
  int b = tid >> 9;
  const float* base = in + (size_t)b * H_ * Q_ + t;
  float s = 0.f, s2 = 0.f;
  for (int c = 0; c < H_; ++c) { float v = base[(size_t)c * Q_]; s += v; s2 += v * v; }
  float mean = s * (1.f / H_);
  float var  = s2 * (1.f / H_) - mean * mean;
  float inv  = rsqrtf(var + 1e-5f);
  for (int c = 0; c < H_; ++c) {
    float v = (base[(size_t)c * Q_] - mean) * inv * g[c] + be[c];
    out[((size_t)b * Q_ + t) * H_ + c] = v;
  }
}

// ---------------------------------------------------------------------------
// GRN on panel-layout z (B, 12, Q, 32):
//   gx[b,c] = ||z[b,c,:]||_2 ; nx = gx / (mean_c gx + 1e-6) ;
//   z = g*(z*nx) + b + z  (in place, bf16)
// grn_gx: wave per (b,kp), lane = ki -> every load is a contiguous 64B line.
// ---------------------------------------------------------------------------
__global__ void grn_gx_kernel(const bf16_t* __restrict__ z, float* __restrict__ gx) {
  int lane = threadIdx.x & 31;
  int grp = (blockIdx.x * blockDim.x + threadIdx.x) >> 5;   // b*12 + kp
  if (grp >= B_ * (HID_ / 32)) return;
  const bf16_t* p = z + ((size_t)grp * Q_) * 32 + lane;
  float s = 0.f;
  for (int t = 0; t < Q_; ++t) { float v = bf2f(p[(size_t)t * 32]); s += v * v; }
  int b = grp / (HID_ / 32), kp = grp % (HID_ / 32);
  gx[(size_t)b * HID_ + kp * 32 + lane] = sqrtf(s);
}

__global__ void grn_nx_kernel(const float* __restrict__ gx, float* __restrict__ nx) {
  __shared__ float sbuf[16];
  int b = blockIdx.x;
  int c = threadIdx.x;            // 384 threads, 12 waves
  int lane = c & 31, wid = c >> 5;
  float g = gx[(size_t)b * HID_ + c];
  float s = g;
  #pragma unroll
  for (int m = 16; m >= 1; m >>= 1) s += __shfl_xor(s, m, 32);
  if (lane == 0) sbuf[wid] = s;
  __syncthreads();
  float tot = 0.f;
  for (int i = 0; i < 12; ++i) tot += sbuf[i];
  float mean = tot * (1.f / HID_);
  nx[(size_t)b * HID_ + c] = g / (mean + 1e-6f);
}

__global__ void grn_apply_kernel(bf16_t* __restrict__ z, const float* __restrict__ nx,
                                 const float* __restrict__ gg, const float* __restrict__ gb) {
  size_t tid = (size_t)blockIdx.x * blockDim.x + threadIdx.x;
  int ki = (int)(tid & 31);
  size_t bkp = tid >> 14;               // b*12 + kp
  int kp = (int)(bkp % (HID_ / 32));
  size_t b = bkp / (HID_ / 32);
  int c = kp * 32 + ki;
  float v = bf2f(z[tid]);
  float out = gg[c] * (v * nx[b * HID_ + c]) + gb[c] + v;
  z[tid] = f2bf(out);
}

// ---------------------------------------------------------------------------
// Heads: from h_seq (B,Q,192) f32 compute kal_rho, kal_phi, gain(B,Q,32)
// Thread per (b,t,j) with j in [0,34): j=0 rho, j=1 phi, else gain[j-2].
// ---------------------------------------------------------------------------
__global__ void heads_kernel(const float* __restrict__ h_seq,
                             const float* __restrict__ rp_w, const float* __restrict__ rp_b,
                             const float* __restrict__ gw, const float* __restrict__ gb,
                             float* __restrict__ krho, float* __restrict__ kphi,
                             float* __restrict__ gain) {
  size_t tid = (size_t)blockIdx.x * blockDim.x + threadIdx.x;
  int j = (int)(tid % 34);
  size_t bt = tid / 34;
  const float* hrow = h_seq + bt * H_;
  const float* wrow;
  float bias;
  if (j < 2) { wrow = rp_w + j * H_; bias = rp_b[j]; }
  else       { wrow = gw + (j - 2) * H_; bias = gb[j - 2]; }
  float s = bias;
  for (int k = 0; k < H_; ++k) s += hrow[k] * wrow[k];
  if (j == 0)      krho[bt] = 1.25f * sigmoidf_(s);
  else if (j == 1) kphi[bt] = PI_F * tanhf(s);
  else             gain[bt * 32 + (j - 2)] = sigmoidf_(s);
}

// ---------------------------------------------------------------------------
// Kalman filter scan (512 sequential steps). Block = one batch row, 16 lanes.
// ---------------------------------------------------------------------------
__global__ void kalman_scan_kernel(const float* __restrict__ x_in,
                                   const float* __restrict__ krho,
                                   const float* __restrict__ kphi,
                                   const float* __restrict__ gain,
                                   float* __restrict__ x_post) {
  int b = blockIdx.x;
  int m = threadIdx.x;
  if (m >= 16) return;
  float re = x_in[(size_t)b * Q_ * D_ + m];
  float im = x_in[(size_t)b * Q_ * D_ + 16 + m];
  for (int t = 0; t < Q_; ++t) {
    size_t bt = (size_t)b * Q_ + t;
    float rho = krho[bt], phi = kphi[bt];
    float sph, cph; sincosf(phi, &sph, &cph);
    float pre = rho * (cph * re - sph * im);
    float pim = rho * (sph * re + cph * im);
    const float* yrow = x_in + bt * D_;
    float gre = gain[bt * 32 + m], gim = gain[bt * 32 + 16 + m];
    re = pre + gre * (yrow[m] - pre);
    im = pim + gim * (yrow[16 + m] - pim);
  }
  x_post[(size_t)b * D_ + m] = re;
  x_post[(size_t)b * D_ + 16 + m] = im;
}

// ---------------------------------------------------------------------------
// Rollout init: h_r = h_seq[:, -1, :], curr = x_post, cat = [bf16(h_r)|bf16(curr)]
// ---------------------------------------------------------------------------
__global__ void roll_init_kernel(const float* __restrict__ h_seq, const float* __restrict__ x_post,
                                 float* __restrict__ h_r, float* __restrict__ curr,
                                 bf16_t* __restrict__ cat) {
  int b = blockIdx.x;
  int c = threadIdx.x;   // 224
  if (c < H_) {
    float v = h_seq[((size_t)b * Q_ + (Q_ - 1)) * H_ + c];
    h_r[(size_t)b * H_ + c] = v;
    cat[(size_t)b * 224 + c] = f2bf(v);
  } else {
    int d = c - H_;
    float v = x_post[(size_t)b * D_ + d];
    curr[(size_t)b * D_ + d] = v;
    cat[(size_t)b * 224 + H_ + d] = f2bf(v);
  }
}

// ---------------------------------------------------------------------------
// GRU gate combine + LayerNorm + rho/phi head. Block = one batch row (192 thr).
// ---------------------------------------------------------------------------
__global__ void gru_update_kernel(const float* __restrict__ gi, const float* __restrict__ gh,
                                  float* __restrict__ h_r,
                                  const float* __restrict__ ln_w, const float* __restrict__ ln_b,
                                  const float* __restrict__ rp_w, const float* __restrict__ rp_b,
                                  bf16_t* __restrict__ cat,
                                  float* __restrict__ rrho, float* __restrict__ rphi) {
  __shared__ float sA[2][8];
  __shared__ float sB[2][8];
  int b = blockIdx.x;
  int c = threadIdx.x;              // 0..191 (6 waves)
  int lane = c & 31, wid = c >> 5;
  const float* gib = gi + (size_t)b * 576;
  const float* ghb = gh + (size_t)b * 576;
  float hprev = h_r[(size_t)b * H_ + c];
  float r = sigmoidf_(gib[c] + ghb[c]);
  float z = sigmoidf_(gib[192 + c] + ghb[192 + c]);
  float n = tanhf(gib[384 + c] + r * ghb[384 + c]);
  float v = (1.f - z) * n + z * hprev;

  float s = v, s2 = v * v;
  #pragma unroll
  for (int m = 16; m >= 1; m >>= 1) { s += __shfl_xor(s, m, 32); s2 += __shfl_xor(s2, m, 32); }
  if (lane == 0) { sA[0][wid] = s; sA[1][wid] = s2; }
  __syncthreads();
  float ts = 0.f, ts2 = 0.f;
  for (int i = 0; i < 6; ++i) { ts += sA[0][i]; ts2 += sA[1][i]; }
  float mean = ts * (1.f / H_);
  float var  = ts2 * (1.f / H_) - mean * mean;
  float inv  = rsqrtf(var + 1e-5f);
  float hn = (v - mean) * inv * ln_w[c] + ln_b[c];
  h_r[(size_t)b * H_ + c] = hn;
  cat[(size_t)b * 224 + c] = f2bf(hn);

  float d0 = hn * rp_w[c];
  float d1 = hn * rp_w[H_ + c];
  #pragma unroll
  for (int m = 16; m >= 1; m >>= 1) { d0 += __shfl_xor(d0, m, 32); d1 += __shfl_xor(d1, m, 32); }
  if (lane == 0) { sB[0][wid] = d0; sB[1][wid] = d1; }
  __syncthreads();
  if (c == 0) {
    float t0 = 0.f, t1 = 0.f;
    for (int i = 0; i < 6; ++i) { t0 += sB[0][i]; t1 += sB[1][i]; }
    rrho[b] = 1.25f * sigmoidf_(t0 + rp_b[0]);
    rphi[b] = PI_F * tanhf(t1 + rp_b[1]);
  }
}

// ---------------------------------------------------------------------------
// Rotate curr with (rho,phi), emit prediction for this step.
// ---------------------------------------------------------------------------
__global__ void rotate_step_kernel(float* __restrict__ curr, const float* __restrict__ rrho,
                                   const float* __restrict__ rphi, bf16_t* __restrict__ cat,
                                   float* __restrict__ preds, int step) {
  int b = blockIdx.x;
  int m = threadIdx.x;
  if (m >= 16) return;
  float rho = rrho[b], phi = rphi[b];
  float sph, cph; sincosf(phi, &sph, &cph);
  float re = curr[(size_t)b * D_ + m];
  float im = curr[(size_t)b * D_ + 16 + m];
  float nre = rho * (cph * re - sph * im);
  float nim = rho * (sph * re + cph * im);
  curr[(size_t)b * D_ + m] = nre;
  curr[(size_t)b * D_ + 16 + m] = nim;
  cat[(size_t)b * 224 + H_ + m] = f2bf(nre);
  cat[(size_t)b * 224 + H_ + 16 + m] = f2bf(nim);
  preds[((size_t)b * WOUT_ + step) * D_ + m] = nre;
  preds[((size_t)b * WOUT_ + step) * D_ + 16 + m] = nim;
}

// ---------------------------------------------------------------------------
// Orchestration
// ---------------------------------------------------------------------------
extern "C" void kernel_launch(void* const* d_in, const int* in_sizes, int n_in,
                              void* d_out, int out_size, void* d_ws, size_t ws_size,
                              hipStream_t stream) {
  (void)in_sizes; (void)n_in; (void)out_size; (void)ws_size;

  const float* x_in      = (const float*)d_in[0];
  const float* inp_w     = (const float*)d_in[1];
  const float* inp_b     = (const float*)d_in[2];
  const float* b_dw_w    = (const float*)d_in[3];
  const float* b_dw_b    = (const float*)d_in[4];
  const float* b_ln_w    = (const float*)d_in[5];
  const float* b_ln_b    = (const float*)d_in[6];
  const float* b_pw1_w   = (const float*)d_in[7];
  const float* b_pw1_b   = (const float*)d_in[8];
  const float* b_grn_g   = (const float*)d_in[9];
  const float* b_grn_b   = (const float*)d_in[10];
  const float* b_pw2_w   = (const float*)d_in[11];
  const float* b_pw2_b   = (const float*)d_in[12];
  const float* out_ln_w  = (const float*)d_in[13];
  const float* out_ln_b  = (const float*)d_in[14];
  const float* fc_rp_w   = (const float*)d_in[15];
  const float* fc_rp_b   = (const float*)d_in[16];
  const float* fc_gain_w = (const float*)d_in[17];
  const float* fc_gain_b = (const float*)d_in[18];
  const float* roll_in_w = (const float*)d_in[19];
  const float* roll_in_b = (const float*)d_in[20];
  const float* gru_wih   = (const float*)d_in[21];
  const float* gru_whh   = (const float*)d_in[22];
  const float* gru_bih   = (const float*)d_in[23];
  const float* gru_bhh   = (const float*)d_in[24];
  const float* roll_ln_w = (const float*)d_in[25];
  const float* roll_ln_b = (const float*)d_in[26];
  const float* fc_rp_r_w = (const float*)d_in[27];
  const float* fc_rp_r_b = (const float*)d_in[28];
  // d_in[29] = w_out (device int); fixed to 64 by setup_inputs.
  float* preds = (float*)d_out;

  char* ws = (char*)d_ws;
  size_t off = 0;
  auto take = [&](size_t bytes) -> char* {
    char* p = ws + off;
    off += (bytes + 255) & ~(size_t)255;
    return p;
  };

  bf16_t* w_inp = (bf16_t*)take((size_t)H_ * INCH_ * 2);
  bf16_t* w_pw1 = (bf16_t*)take((size_t)2 * HID_ * H_ * 2);
  bf16_t* w_pw2 = (bf16_t*)take((size_t)2 * H_ * HID_ * 2);
  bf16_t* w_rin = (bf16_t*)take((size_t)H_ * 224 * 2);
  bf16_t* w_wih = (bf16_t*)take((size_t)576 * H_ * 2);
  bf16_t* w_whh = (bf16_t*)take((size_t)576 * H_ * 2);
  float*  gain  = (float*)take((size_t)B_ * Q_ * 32 * 4);
  float*  krho  = (float*)take((size_t)B_ * Q_ * 4);
  float*  kphi  = (float*)take((size_t)B_ * Q_ * 4);
  float*  gx    = (float*)take((size_t)B_ * HID_ * 4);
  float*  nx    = (float*)take((size_t)B_ * HID_ * 4);
  float*  x_post= (float*)take((size_t)B_ * D_ * 4);
  float*  h_r   = (float*)take((size_t)B_ * H_ * 4);
  bf16_t* cat   = (bf16_t*)take((size_t)B_ * 224 * 2);
  bf16_t* x_bf  = (bf16_t*)take((size_t)B_ * H_ * 2);
  float*  gi    = (float*)take((size_t)B_ * 576 * 4);
  float*  gh    = (float*)take((size_t)B_ * 576 * 4);
  float*  curr  = (float*)take((size_t)B_ * D_ * 4);
  float*  rrho  = (float*)take((size_t)B_ * 4);
  float*  rphi  = (float*)take((size_t)B_ * 4);
  bf16_t* feats = (bf16_t*)take((size_t)B_ * INCH_ * Q_ * 2);   // panels (B,3,Q,32)
  float*  hbuf  = (float*)take((size_t)B_ * H_ * Q_ * 4);
  float*  ydw   = (float*)take((size_t)B_ * H_ * Q_ * 4);       // reused as h_seq
  float*  h_seq = ydw;
  bf16_t* yln   = (bf16_t*)take((size_t)B_ * H_ * Q_ * 2);      // panels (B,6,Q,32)
  bf16_t* zbuf  = (bf16_t*)take((size_t)B_ * HID_ * Q_ * 2);    // panels (B,12,Q,32)

  auto cvt = [&](const float* s, bf16_t* d, int n) {
    f32_to_bf16_kernel<<<(n + 255) / 256, 256, 0, stream>>>(s, d, n);
  };
  cvt(inp_w,     w_inp, H_ * INCH_);
  cvt(b_pw1_w,   w_pw1, 2 * HID_ * H_);
  cvt(b_pw2_w,   w_pw2, 2 * H_ * HID_);
  cvt(roll_in_w, w_rin, H_ * 224);
  cvt(gru_wih,   w_wih, 576 * H_);
  cvt(gru_whh,   w_whh, 576 * H_);

  // Features + input projection (K=96)
  features_kernel<<<(B_ * D_ * Q_) / 256, 256, 0, stream>>>(x_in, feats);
  gemm_wx_kernel<INCH_, Q_, 0, false, true, false>
      <<<dim3(Q_ / 128, H_ / 16, B_), 128, 0, stream>>>(
          w_inp, feats, inp_b, nullptr, hbuf, nullptr, H_);

  // ConvNeXt blocks
  for (int i = 0; i < 2; ++i) {
    dwconv_kernel<<<(B_ * H_ * Q_) / 256, 256, 0, stream>>>(
        hbuf, b_dw_w + (size_t)i * H_ * KER_, b_dw_b + (size_t)i * H_, ydw);
    ln_ch_bf_kernel<<<(B_ * Q_) / 256, 256, 0, stream>>>(
        ydw, yln, b_ln_w + (size_t)i * H_, b_ln_b + (size_t)i * H_);
    gemm_wx_kernel<H_, Q_, 1, false, false, true>             // pw1 + GELU (K=192)
        <<<dim3(Q_ / 128, HID_ / 16, B_), 128, 0, stream>>>(
            w_pw1 + (size_t)i * HID_ * H_, yln, b_pw1_b + (size_t)i * HID_,
            nullptr, nullptr, zbuf, HID_);
    grn_gx_kernel<<<(B_ * (HID_ / 32) * 32) / 256, 256, 0, stream>>>(zbuf, gx);
    grn_nx_kernel<<<B_, HID_, 0, stream>>>(gx, nx);
    grn_apply_kernel<<<(int)(((size_t)B_ * HID_ * Q_) / 256), 256, 0, stream>>>(
        zbuf, nx, b_grn_g + (size_t)i * HID_, b_grn_b + (size_t)i * HID_);
    gemm_wx_kernel<HID_, Q_, 0, true, true, false>            // pw2 + residual (K=384)
        <<<dim3(Q_ / 128, H_ / 16, B_), 128, 0, stream>>>(
            w_pw2 + (size_t)i * H_ * HID_, zbuf, b_pw2_b + (size_t)i * H_,
            hbuf, hbuf, nullptr, H_);
  }

  // Output LN (transposed to (B,Q,H)) + heads + Kalman scan
  ln_ch_t_kernel<<<(B_ * Q_) / 256, 256, 0, stream>>>(hbuf, h_seq, out_ln_w, out_ln_b);
  heads_kernel<<<(int)(((size_t)B_ * Q_ * 34) / 256), 256, 0, stream>>>(
      h_seq, fc_rp_w, fc_rp_b, fc_gain_w, fc_gain_b, krho, kphi, gain);
  kalman_scan_kernel<<<B_, 32, 0, stream>>>(x_in, krho, kphi, gain, x_post);
  roll_init_kernel<<<B_, 224, 0, stream>>>(h_seq, x_post, h_r, curr, cat);

  // GRU rollout (64 sequential steps; each step = 3 WMMA GEMMs + gates + rotate)
  for (int s = 0; s < WOUT_; ++s) {
    gemm_bt_kernel<224, H_, 1, false, true>                   // x = tanh([h|curr] Wr^T)
        <<<dim3(H_ / 64, B_ / 32), 128, 0, stream>>>(
            cat, 224, w_rin, roll_in_b, nullptr, x_bf);
    gemm_bt_kernel<H_, 576, 0, true, false>                   // gi = x Wih^T + bih
        <<<dim3(576 / 64, B_ / 32), 128, 0, stream>>>(
            x_bf, H_, w_wih, gru_bih, gi, nullptr);
    gemm_bt_kernel<H_, 576, 0, true, false>                   // gh = h Whh^T + bhh
        <<<dim3(576 / 64, B_ / 32), 128, 0, stream>>>(
            cat, 224, w_whh, gru_bhh, gh, nullptr);
    gru_update_kernel<<<B_, H_, 0, stream>>>(
        gi, gh, h_r, roll_ln_w, roll_ln_b, fc_rp_r_w, fc_rp_r_b, cat, rrho, rphi);
    rotate_step_kernel<<<B_, 32, 0, stream>>>(curr, rrho, rphi, cat, preds, s);
  }
}
